// SimpleTransformerBlock_30313879175853
// MI455X (gfx1250) — compile-verified, matrix operand order checked by reading, not run on hardware
//
#include <hip/hip_runtime.h>
#include <hip/hip_bf16.h>

// ---------------------------------------------------------------------------
// Types for CDNA5 WMMA (wave32, 16x16x32 bf16 -> f32)
// ---------------------------------------------------------------------------
typedef __bf16 bf16_t;
typedef __attribute__((ext_vector_type(16))) __bf16 v16bf;
typedef __attribute__((ext_vector_type(8)))  __bf16 v8bf;
typedef __attribute__((ext_vector_type(8)))  float  v8f;

union FragU { v16bf v; v8bf h[2]; };

// ---------------------------------------------------------------------------
// CDNA5 async global->LDS copy (GLOBAL_LOAD_ASYNC_TO_LDS_B128, ASYNCcnt).
// Inline asm: VDST = VGPR with LDS byte offset; VADDR = 64-bit global address.
// ---------------------------------------------------------------------------
__device__ __forceinline__ void async_ld_b128(void* lds_ptr, const void* gptr) {
    const unsigned lds_off = (unsigned)(unsigned long long)(uintptr_t)lds_ptr;
    const unsigned long long ga = (unsigned long long)(uintptr_t)gptr;
    asm volatile("global_load_async_to_lds_b128 %0, %1, off"
                 :: "v"(lds_off), "v"(ga)
                 : "memory");
}

template<int N>
__device__ __forceinline__ void wait_asynccnt() {
#if __has_builtin(__builtin_amdgcn_s_wait_asynccnt)
    __builtin_amdgcn_s_wait_asynccnt(N);
#else
    if constexpr (N == 0) asm volatile("s_wait_asynccnt 0x0" ::: "memory");
    else                  asm volatile("s_wait_asynccnt 0x4" ::: "memory");
#endif
}

enum {
    EPI_BIAS_BF16 = 0,   // C_bf16 = acc + bias[n]
    EPI_F32,             // C_f32  = acc                       (batched)
    EPI_BF16,            // C_bf16 = acc                       (batched)
    EPI_BIAS_RES_F32,    // C_f32  = acc + bias[n] + res[m,n]
    EPI_SIGMOID_BF16,    // C_bf16 = sigmoid(acc + bias[n])
    EPI_MUL_BF16         // C_bf16 = (acc + bias[n]) * extra_bf16[m,n]
};

#define BM 128
#define BN 128
#define BK 32

// ---------------------------------------------------------------------------
// Tiled bf16 GEMM:  C[m,n] = sum_k A[m,k] * W[n,k]   (A: MxK, W: NxK row-major)
// 256 threads = 8 waves in a 2(M) x 4(N) grid; each wave owns 64x32 of C as a
// 4x2 grid of 16x16 WMMA tiles. Global->LDS staging is double-buffered via
// GLOBAL_LOAD_ASYNC_TO_LDS_B128: tile k+1 streams while tile k computes; since
// per-wave async loads retire in order, s_wait_asynccnt 4 == "tile k done".
// ---------------------------------------------------------------------------
template<int EPI>
__global__ __launch_bounds__(256) void gemm_bf16_wmma(
    const bf16_t* __restrict__ A, long long aBatch,
    const bf16_t* __restrict__ W, long long wBatch,
    const float*  __restrict__ bias,
    const void*   __restrict__ extra,
    void*         __restrict__ Cout, long long cBatch,
    int M, int N, int K)
{
    __shared__ bf16_t As[2][BM][BK];
    __shared__ bf16_t Bs[2][BN][BK];

    const int z = blockIdx.z;
    A += (long long)z * aBatch;
    W += (long long)z * wBatch;

    const int tid   = threadIdx.x;
    const int lane  = tid & 31;
    const int wave  = tid >> 5;
    const int wm    = wave >> 2;           // 0..1 (M direction)
    const int wn    = wave & 3;            // 0..3 (N direction)
    const int bm    = blockIdx.y * BM;
    const int bn    = blockIdx.x * BN;
    const int lrow  = lane & 15;           // row (A) / col (B) within 16
    const int khalf = (lane >> 4) * 8;     // K sub-offset per ISA bf16 layout

    v8f acc[4][2];
#pragma unroll
    for (int i = 0; i < 4; ++i)
#pragma unroll
        for (int j = 0; j < 2; ++j)
            acc[i][j] = (v8f){0.f,0.f,0.f,0.f,0.f,0.f,0.f,0.f};

    const int ldrow = tid >> 2;            // 0..63
    const int ldk   = (tid & 3) * 8;       // 0,8,16,24 (bf16 elems, 16B chunks)

    // Issue one tile's 4 async 16B copies (2 rows of A, 2 rows of W per lane).
    auto issue_tile = [&](int kk, int b) {
#pragma unroll
        for (int i = 0; i < 2; ++i) {
            const int row = ldrow + 64 * i;
            async_ld_b128(&As[b][row][ldk],
                          &A[(long long)(bm + row) * K + kk + ldk]);
            async_ld_b128(&Bs[b][row][ldk],
                          &W[(long long)(bn + row) * K + kk + ldk]);
        }
    };

    issue_tile(0, 0);            // prologue: tile 0 -> buffer 0
    int buf = 0;

    for (int k0 = 0; k0 < K; k0 += BK) {
        const bool more = (k0 + BK) < K;
        if (more) {
            // buf^1 was released by the trailing barrier of the previous
            // iteration; stream tile k+1 into it while tile k computes.
            issue_tile(k0 + BK, buf ^ 1);
            __builtin_prefetch(&A[(long long)(bm + ldrow) * K + k0 + 2 * BK + ldk], 0, 0);
            __builtin_prefetch(&W[(long long)(bn + ldrow) * K + k0 + 2 * BK + ldk], 0, 0);
            wait_asynccnt<4>();  // tile k's 4 copies retired (in-order)
        } else {
            wait_asynccnt<0>();
        }
        __syncthreads();

        // Per-lane fragments (CDNA5 16-bit 16x32 layout: lanes 0-15 K 0-7/16-23,
        // lanes 16-31 K 8-15/24-31; identical addressing for A rows and B cols).
        FragU a[4], b[2];
#pragma unroll
        for (int mi = 0; mi < 4; ++mi) {
            const int r = wm * 64 + mi * 16 + lrow;
            a[mi].h[0] = *(const v8bf*)&As[buf][r][khalf];
            a[mi].h[1] = *(const v8bf*)&As[buf][r][khalf + 16];
        }
#pragma unroll
        for (int ni = 0; ni < 2; ++ni) {
            const int c = wn * 32 + ni * 16 + lrow;
            b[ni].h[0] = *(const v8bf*)&Bs[buf][c][khalf];
            b[ni].h[1] = *(const v8bf*)&Bs[buf][c][khalf + 16];
        }
#pragma unroll
        for (int mi = 0; mi < 4; ++mi)
#pragma unroll
            for (int ni = 0; ni < 2; ++ni)
                acc[mi][ni] = __builtin_amdgcn_wmma_f32_16x16x32_bf16(
                    false, a[mi].v, false, b[ni].v,
                    (short)0, acc[mi][ni], false, false);

        __syncthreads();         // all waves done reading buf -> reusable
        buf ^= 1;
    }

    // Epilogue. 16x16 f32 C layout: VGPR r -> M = r + 8*(lane>>4), N = lane&15.
#pragma unroll
    for (int mi = 0; mi < 4; ++mi) {
#pragma unroll
        for (int ni = 0; ni < 2; ++ni) {
            const int colg = bn + wn * 32 + ni * 16 + lrow;
#pragma unroll
            for (int r = 0; r < 8; ++r) {
                const int rowg = bm + wm * 64 + mi * 16 + (lane >> 4) * 8 + r;
                const long long idx = (long long)rowg * N + colg;
                const float v = acc[mi][ni][r];
                if constexpr (EPI == EPI_BIAS_BF16) {
                    bf16_t* cb = (bf16_t*)Cout + (long long)z * cBatch;
                    cb[idx] = (bf16_t)(v + bias[colg]);
                } else if constexpr (EPI == EPI_F32) {
                    float* cf = (float*)Cout + (long long)z * cBatch;
                    cf[idx] = v;
                } else if constexpr (EPI == EPI_BF16) {
                    bf16_t* cb = (bf16_t*)Cout + (long long)z * cBatch;
                    cb[idx] = (bf16_t)v;
                } else if constexpr (EPI == EPI_BIAS_RES_F32) {
                    const float* res = (const float*)extra;
                    ((float*)Cout)[idx] = v + bias[colg] + res[idx];
                } else if constexpr (EPI == EPI_SIGMOID_BF16) {
                    const float t = v + bias[colg];
                    ((bf16_t*)Cout)[idx] = (bf16_t)(1.0f / (1.0f + __expf(-t)));
                } else { // EPI_MUL_BF16
                    const bf16_t* sg = (const bf16_t*)extra;
                    ((bf16_t*)Cout)[idx] = (bf16_t)((v + bias[colg]) * (float)sg[idx]);
                }
            }
        }
    }
}

// ---------------------------------------------------------------------------
// LayerNorm over rows of 1024 f32 -> bf16
// ---------------------------------------------------------------------------
__global__ __launch_bounds__(256) void layernorm_bf16_k(
    const float* __restrict__ x, const float* __restrict__ g,
    const float* __restrict__ be, bf16_t* __restrict__ out)
{
    __shared__ float r1[256], r2[256];
    const long long row = blockIdx.x;
    const float4 v = *(const float4*)(x + row * 1024 + threadIdx.x * 4);
    r1[threadIdx.x] = v.x + v.y + v.z + v.w;
    r2[threadIdx.x] = v.x*v.x + v.y*v.y + v.z*v.z + v.w*v.w;
    __syncthreads();
    for (int off = 128; off; off >>= 1) {
        if (threadIdx.x < off) {
            r1[threadIdx.x] += r1[threadIdx.x + off];
            r2[threadIdx.x] += r2[threadIdx.x + off];
        }
        __syncthreads();
    }
    const float mu   = r1[0] * (1.0f / 1024.0f);
    const float var  = r2[0] * (1.0f / 1024.0f) - mu * mu;
    const float rstd = rsqrtf(var + 1e-5f);
    const int c = threadIdx.x * 4;
    bf16_t* o = out + row * 1024 + c;
    o[0] = (bf16_t)((v.x - mu) * rstd * g[c+0] + be[c+0]);
    o[1] = (bf16_t)((v.y - mu) * rstd * g[c+1] + be[c+1]);
    o[2] = (bf16_t)((v.z - mu) * rstd * g[c+2] + be[c+2]);
    o[3] = (bf16_t)((v.w - mu) * rstd * g[c+3] + be[c+3]);
}

// ---------------------------------------------------------------------------
// Row softmax over 2048 f32 scores (pre-scaled by 1/sqrt(1024)) -> bf16
// ---------------------------------------------------------------------------
__global__ __launch_bounds__(256) void softmax_row_bf16(
    const float* __restrict__ scores, bf16_t* __restrict__ attn)
{
    __shared__ float red[256];
    const long long row = blockIdx.x;
    const float* src = scores + row * 2048;
    float v[8];
    float mx = -3.4e38f;
#pragma unroll
    for (int i = 0; i < 8; ++i) {
        v[i] = src[threadIdx.x + 256 * i] * 0.03125f; // 1/sqrt(1024)
        mx = fmaxf(mx, v[i]);
    }
    red[threadIdx.x] = mx; __syncthreads();
    for (int off = 128; off; off >>= 1) {
        if (threadIdx.x < off)
            red[threadIdx.x] = fmaxf(red[threadIdx.x], red[threadIdx.x + off]);
        __syncthreads();
    }
    mx = red[0];
    __syncthreads();
    float s = 0.f;
#pragma unroll
    for (int i = 0; i < 8; ++i) { v[i] = __expf(v[i] - mx); s += v[i]; }
    red[threadIdx.x] = s; __syncthreads();
    for (int off = 128; off; off >>= 1) {
        if (threadIdx.x < off) red[threadIdx.x] += red[threadIdx.x + off];
        __syncthreads();
    }
    const float inv = 1.0f / red[0];
    bf16_t* dst = attn + row * 2048;
#pragma unroll
    for (int i = 0; i < 8; ++i)
        dst[threadIdx.x + 256 * i] = (bf16_t)(v[i] * inv);
}

// ---------------------------------------------------------------------------
// Batched bf16 transpose: in [R x C] -> out [C x R], per batch z
// ---------------------------------------------------------------------------
__global__ __launch_bounds__(1024) void transpose_bf16_k(
    const bf16_t* __restrict__ in, bf16_t* __restrict__ out, int R, int C)
{
    __shared__ bf16_t t[32][33];
    const long long z = blockIdx.z;
    in  += z * (long long)R * C;
    out += z * (long long)R * C;
    const int c = blockIdx.x * 32 + threadIdx.x;
    const int r = blockIdx.y * 32 + threadIdx.y;
    t[threadIdx.y][threadIdx.x] = in[(long long)r * C + c];
    __syncthreads();
    const int orow = blockIdx.x * 32 + threadIdx.y;   // along C
    const int ocol = blockIdx.y * 32 + threadIdx.x;   // along R
    out[(long long)orow * R + ocol] = t[threadIdx.x][threadIdx.y];
}

// ---------------------------------------------------------------------------
// f32 -> bf16 conversion (weights), n multiple of 4
// ---------------------------------------------------------------------------
__global__ __launch_bounds__(256) void cvt_f32_bf16(
    const float* __restrict__ in, bf16_t* __restrict__ out, long long n)
{
    const long long i = ((long long)blockIdx.x * blockDim.x + threadIdx.x) * 4;
    if (i + 3 < n) {
        const float4 v = *(const float4*)(in + i);
        out[i + 0] = (bf16_t)v.x;
        out[i + 1] = (bf16_t)v.y;
        out[i + 2] = (bf16_t)v.z;
        out[i + 3] = (bf16_t)v.w;
    }
}

// ---------------------------------------------------------------------------
// Host-side orchestration
// ---------------------------------------------------------------------------
extern "C" void kernel_launch(void* const* d_in, const int* in_sizes, int n_in,
                              void* d_out, int out_size, void* d_ws, size_t ws_size,
                              hipStream_t stream)
{
    (void)in_sizes; (void)n_in; (void)out_size; (void)ws_size;

    const float* x      = (const float*)d_in[0];
    const float* wq     = (const float*)d_in[1];
    const float* bq     = (const float*)d_in[2];
    const float* wk     = (const float*)d_in[3];
    const float* bk     = (const float*)d_in[4];
    const float* wv     = (const float*)d_in[5];
    const float* bv     = (const float*)d_in[6];
    const float* wo     = (const float*)d_in[7];
    const float* bo     = (const float*)d_in[8];
    const float* w_up   = (const float*)d_in[9];
    const float* b_up   = (const float*)d_in[10];
    const float* w_gate = (const float*)d_in[11];
    const float* b_gate = (const float*)d_in[12];
    const float* w_down = (const float*)d_in[13];
    const float* b_down = (const float*)d_in[14];
    const float* g1     = (const float*)d_in[15];
    const float* be1    = (const float*)d_in[16];
    const float* g2     = (const float*)d_in[17];
    const float* be2    = (const float*)d_in[18];

    constexpr int Dm = 1024, FFd = 4096, S = 2048, B = 4;
    constexpr int M  = B * S;                       // 8192 tokens
    constexpr long long SD = (long long)S * Dm;
    constexpr long long SS = (long long)S * S;

    size_t cur = 0;
    auto alloc = [&](size_t bytes) -> void* {
        void* p = (char*)d_ws + cur;
        cur += (bytes + 255) & ~(size_t)255;
        return p;
    };

    bf16_t* wqb    = (bf16_t*)alloc((size_t)Dm * Dm  * 2);
    bf16_t* wkb    = (bf16_t*)alloc((size_t)Dm * Dm  * 2);
    bf16_t* wvb    = (bf16_t*)alloc((size_t)Dm * Dm  * 2);
    bf16_t* wob    = (bf16_t*)alloc((size_t)Dm * Dm  * 2);
    bf16_t* wupb   = (bf16_t*)alloc((size_t)FFd * Dm * 2);
    bf16_t* wgateb = (bf16_t*)alloc((size_t)FFd * Dm * 2);
    bf16_t* wdownb = (bf16_t*)alloc((size_t)Dm * FFd * 2);
    bf16_t* h1     = (bf16_t*)alloc((size_t)M * Dm   * 2);
    bf16_t* qb     = (bf16_t*)alloc((size_t)M * Dm   * 2);
    bf16_t* kb     = (bf16_t*)alloc((size_t)M * Dm   * 2);
    bf16_t* vb     = (bf16_t*)alloc((size_t)M * Dm   * 2);
    bf16_t* vT     = (bf16_t*)alloc((size_t)M * Dm   * 2);
    float*  scores = (float*) alloc((size_t)B * SS   * 4);
    bf16_t* attn   = (bf16_t*)alloc((size_t)B * SS   * 2);
    bf16_t* aout   = (bf16_t*)alloc((size_t)M * Dm   * 2);
    float*  x1     = (float*) alloc((size_t)M * Dm   * 4);
    bf16_t* h2     = (bf16_t*)alloc((size_t)M * Dm   * 2);
    bf16_t* sgate  = (bf16_t*)alloc((size_t)M * FFd  * 2);
    bf16_t* ff     = (bf16_t*)alloc((size_t)M * FFd  * 2);

    // --- weight conversions (every call; no caching) ---
    {
        const float* src[7] = { wq, wk, wv, wo, w_up, w_gate, w_down };
        bf16_t* dst[7]      = { wqb, wkb, wvb, wob, wupb, wgateb, wdownb };
        const long long n7[7] = {
            (long long)Dm*Dm, (long long)Dm*Dm, (long long)Dm*Dm, (long long)Dm*Dm,
            (long long)FFd*Dm, (long long)FFd*Dm, (long long)Dm*FFd };
        for (int i = 0; i < 7; ++i) {
            const int grid = (int)(n7[i] / 1024);
            cvt_f32_bf16<<<grid, 256, 0, stream>>>(src[i], dst[i], n7[i]);
        }
    }

    // --- LN1 -> h1 (bf16) ---
    layernorm_bf16_k<<<M, 256, 0, stream>>>(x, g1, be1, h1);

    // --- QKV projections ---
    const dim3 gProjD(Dm / BN, M / BM, 1);
    gemm_bf16_wmma<EPI_BIAS_BF16><<<gProjD, 256, 0, stream>>>(
        h1, 0, wqb, 0, bq, nullptr, qb, 0, M, Dm, Dm);
    gemm_bf16_wmma<EPI_BIAS_BF16><<<gProjD, 256, 0, stream>>>(
        h1, 0, wkb, 0, bk, nullptr, kb, 0, M, Dm, Dm);
    gemm_bf16_wmma<EPI_BIAS_BF16><<<gProjD, 256, 0, stream>>>(
        h1, 0, wvb, 0, bv, nullptr, vb, 0, M, Dm, Dm);

    // --- scores = q @ k^T (batched over B) ---
    const dim3 gScores(S / BN, S / BM, B);
    gemm_bf16_wmma<EPI_F32><<<gScores, 256, 0, stream>>>(
        qb, SD, kb, SD, nullptr, nullptr, scores, SS, S, S, Dm);

    // --- softmax (scale 1/32) -> attn bf16 ---
    softmax_row_bf16<<<B * S, 256, 0, stream>>>(scores, attn);

    // --- v^T per batch: [S x D] -> [D x S] ---
    transpose_bf16_k<<<dim3(Dm / 32, S / 32, B), dim3(32, 32), 0, stream>>>(
        vb, vT, S, Dm);

    // --- attn_out = attn @ v  (== attn @ (vT)^T) ---
    const dim3 gAOut(Dm / BN, S / BM, B);
    gemm_bf16_wmma<EPI_BF16><<<gAOut, 256, 0, stream>>>(
        attn, SS, vT, SD, nullptr, nullptr, aout, SD, S, Dm, S);

    // --- O projection + residual -> x1 (f32) ---
    gemm_bf16_wmma<EPI_BIAS_RES_F32><<<gProjD, 256, 0, stream>>>(
        aout, 0, wob, 0, bo, x, x1, 0, M, Dm, Dm);

    // --- LN2 -> h2 (bf16) ---
    layernorm_bf16_k<<<M, 256, 0, stream>>>(x1, g2, be2, h2);

    // --- gate: sigmoid(h2 @ w_gate^T + b_gate) -> sgate bf16 ---
    const dim3 gFF(FFd / BN, M / BM, 1);
    gemm_bf16_wmma<EPI_SIGMOID_BF16><<<gFF, 256, 0, stream>>>(
        h2, 0, wgateb, 0, b_gate, nullptr, sgate, 0, M, FFd, Dm);

    // --- up: (h2 @ w_up^T + b_up) * sgate -> ff bf16 ---
    gemm_bf16_wmma<EPI_MUL_BF16><<<gFF, 256, 0, stream>>>(
        h2, 0, wupb, 0, b_up, sgate, ff, 0, M, FFd, Dm);

    // --- down: ff @ w_down^T + b_down + x1 -> d_out f32 ---
    gemm_bf16_wmma<EPI_BIAS_RES_F32><<<gProjD, 256, 0, stream>>>(
        ff, 0, wdownb, 0, b_down, x1, (float*)d_out, 0, M, Dm, FFd);
}
